// Decoder_10419590660720
// MI455X (gfx1250) — matrix operand order repaired and checked by reading, not compiled, not run on hardware
//
#include <hip/hip_runtime.h>
#include <hip/hip_bf16.h>

// Sizes from reference
#define VSZ 32000
#define ESZ 256
#define HSZ 256
#define BSZ 32
#define TSZ 64
#define TB  (TSZ*BSZ)      // 2048 rows
#define G3H (3*HSZ)        // 768
#define OUTV (BSZ*TSZ*VSZ) // 65,536,000

typedef __attribute__((ext_vector_type(16))) _Float16 v16h;
typedef __attribute__((ext_vector_type(8)))  _Float16 v8h;
typedef __attribute__((ext_vector_type(8)))  float    v8f;

// ---- WMMA helpers (CDNA5 wave32, 16x16x32 f16 -> f32) --------------------

__device__ __forceinline__ v8f wmma_f16(v16h a, v16h b, v8f c) {
  // D = A(16x32) * B(32x16) + C(16x16 f32)
  return __builtin_amdgcn_wmma_f32_16x16x32_f16(
      /*neg_a=*/false, a, /*neg_b=*/false, b,
      /*c_mod=*/(short)0, c, /*reuse_a=*/false, /*reuse_b=*/false);
}

// Load A tile (16 rows x 32 K) from row-major f16, row stride `ld` elements.
// ISA layout: lanes 0-15 row M=lane, K={0..7,16..23}; lanes 16-31 K={8..15,24..31}.
__device__ __forceinline__ v16h load_a_tile(const _Float16* base, int ld) {
  int lane = threadIdx.x & 31;
  int row  = lane & 15;
  int kb   = (lane >> 4) << 3;                 // 0 or 8
  const _Float16* p = base + row * ld + kb;
  v8h lo = *(const v8h*)(p);                   // K = kb..kb+7
  v8h hi = *(const v8h*)(p + 16);              // K = kb+16..kb+23
  v16h a;
#pragma unroll
  for (int i = 0; i < 8; ++i) { a[i] = lo[i]; a[i + 8] = hi[i]; }
  return a;
}

// Load B tile (32 K x 16 N) where B[k][n] = W[n0+n][k0+k], W row-major [N][K]
// with stride `ld`. ISA layout: lanes 0-15 col N=lane K={0..7,16..23}; lanes 16-31 K={8..15,24..31}.
// Each lane: two runs of 8 contiguous f16 -> coalesced-friendly.
__device__ __forceinline__ v16h load_b_tile(const _Float16* base, int ld) {
  int lane = threadIdx.x & 31;
  int n    = lane & 15;
  int kb   = (lane >> 4) << 3;                 // 0 or 8
  const _Float16* p = base + n * ld + kb;
  v8h lo = *(const v8h*)(p);
  v8h hi = *(const v8h*)(p + 16);
  v16h b;
#pragma unroll
  for (int i = 0; i < 8; ++i) { b[i] = lo[i]; b[i + 8] = hi[i]; }
  return b;
}

// ---- small utility kernels ------------------------------------------------

__global__ void k_cvt_f32_f16(const float* __restrict__ src,
                              _Float16* __restrict__ dst, int n) {
  int i = blockIdx.x * blockDim.x + threadIdx.x;
  int stride = gridDim.x * blockDim.x;
  for (; i < n; i += stride) dst[i] = (_Float16)src[i];
}

// Gather teacher-forced embeddings: row = t*B + b, token = t==0 ? SOS : target[b][t-1]
__global__ void k_embed(const int* __restrict__ target, const float* __restrict__ emb,
                        _Float16* __restrict__ Xf) {
  int row = blockIdx.x;                 // 0..2047
  int t = row >> 5, b = row & 31;
  int tok = (t == 0) ? 1 : target[b * TSZ + (t - 1)];
  Xf[row * ESZ + threadIdx.x] = (_Float16)emb[tok * ESZ + threadIdx.x];
}

// ---- GX = X @ W_ih^T + b_ih  (2048 x 768, K=256) --------------------------
// One block per 16-row M tile; 8 waves each own 6 N tiles. A register-resident.
__global__ __launch_bounds__(256) void k_gx(const _Float16* __restrict__ Xf,
                                            const _Float16* __restrict__ Wih,
                                            const float* __restrict__ bih,
                                            float* __restrict__ GX) {
  int wave = threadIdx.x >> 5, lane = threadIdx.x & 31;
  int m0 = blockIdx.x * 16;
  int mo = (lane >> 4) * 8, n = lane & 15;
  v16h A[8];
#pragma unroll
  for (int k = 0; k < 8; ++k) A[k] = load_a_tile(Xf + m0 * ESZ + k * 32, ESZ);
  for (int j = 0; j < 6; ++j) {
    int nt = wave * 6 + j;
    v8f acc = {};
#pragma unroll
    for (int k = 0; k < 8; ++k) {
      v16h Bt = load_b_tile(Wih + nt * 16 * ESZ + k * 32, ESZ);
      acc = wmma_f16(A[k], Bt, acc);
    }
    int col = nt * 16 + n;
    float bias = bih[col];
#pragma unroll
    for (int r = 0; r < 8; ++r)
      GX[(m0 + mo + r) * G3H + col] = acc[r] + bias;
  }
}

// ---- Sequential GRU recurrence (single workgroup, 64 steps) ---------------
// Per step: gh = h @ W_hh^T via WMMA (A tiles from LDS), then gate math,
// h written back to LDS (f16) + H_all (f16). W_hh stays L2-hot (384 KB f16).
__global__ __launch_bounds__(256) void k_gru(const float* __restrict__ h0,
                                             const _Float16* __restrict__ Whh,
                                             const float* __restrict__ bhh,
                                             const float* __restrict__ GX,
                                             _Float16* __restrict__ Hall,
                                             float* __restrict__ hidden_out) {
  __shared__ _Float16 hs[BSZ * HSZ];     // 16 KB
  __shared__ _Float16 gh[BSZ * G3H];     // 48 KB
  int tid = threadIdx.x;
  int wave = tid >> 5, lane = tid & 31;
  int mo = (lane >> 4) * 8, n = lane & 15;

  for (int i = tid; i < BSZ * HSZ; i += 256) hs[i] = (_Float16)h0[i];
  __syncthreads();

  for (int t = 0; t < TSZ; ++t) {
    // gh = hs @ Whh^T   (M=32 -> 2 tiles, N=768 -> 48 tiles, K=256 -> 8 steps)
    for (int mt = 0; mt < 2; ++mt) {
      v16h A[8];
#pragma unroll
      for (int k = 0; k < 8; ++k)
        A[k] = load_a_tile(hs + mt * 16 * HSZ + k * 32, HSZ);
      for (int j = 0; j < 6; ++j) {
        int nt = wave * 6 + j;
        v8f acc = {};
#pragma unroll
        for (int k = 0; k < 8; ++k) {
          v16h Bt = load_b_tile(Whh + nt * 16 * HSZ + k * 32, HSZ);
          acc = wmma_f16(A[k], Bt, acc);
        }
        int col = nt * 16 + n;
#pragma unroll
        for (int r = 0; r < 8; ++r)
          gh[(mt * 16 + mo + r) * G3H + col] = (_Float16)acc[r];
      }
    }
    __syncthreads();

    // Gate math: each thread owns 32 (b,j) elements exclusively.
    for (int i = 0; i < 32; ++i) {
      int e = tid + i * 256;                 // 0..8191
      int b = e >> 8, j = e & 255;
      int row = t * BSZ + b;
      float xr = GX[row * G3H + j];
      float xz = GX[row * G3H + 256 + j];
      float xn = GX[row * G3H + 512 + j];
      float hr = (float)gh[b * G3H + j]        + bhh[j];
      float hz = (float)gh[b * G3H + 256 + j]  + bhh[256 + j];
      float hn = (float)gh[b * G3H + 512 + j]  + bhh[512 + j];
      float r  = 1.0f / (1.0f + __expf(-(xr + hr)));
      float z  = 1.0f / (1.0f + __expf(-(xz + hz)));
      float nn = tanhf(xn + r * hn);
      float hp = (float)hs[b * HSZ + j];
      float hv = (1.0f - z) * nn + z * hp;
      hs[b * HSZ + j] = (_Float16)hv;
      Hall[row * HSZ + j] = (_Float16)hv;
      if (t == TSZ - 1) hidden_out[b * HSZ + j] = hv;
    }
    __syncthreads();
  }
}

// ---- Projection + fused log_softmax stats ---------------------------------
// Block = one 16-row M tile (rows are t*B+b). 8 waves stride over the 2000
// N tiles. Pass A: WMMA GEMM + bias, write raw logits, online per-row max.
// Cross-lane (width-16) + cross-wave reductions, then pass B re-reads the
// L2-hot logits to accumulate sum(exp(x - max)) -> lse[row] = max + log(sum).
__global__ __launch_bounds__(256) void k_proj(const _Float16* __restrict__ Hall,
                                              const _Float16* __restrict__ Wfc,
                                              const float* __restrict__ bfc,
                                              float* __restrict__ out,
                                              float* __restrict__ lse) {
  __shared__ float red[8][16];
  __shared__ float rowstat[16];
  int wave = threadIdx.x >> 5, lane = threadIdx.x & 31;
  int m0 = blockIdx.x * 16;
  int mo = (lane >> 4) * 8, n = lane & 15;

  v16h A[8];
#pragma unroll
  for (int k = 0; k < 8; ++k) A[k] = load_a_tile(Hall + m0 * HSZ + k * 32, HSZ);

  int obase[8];                                // out is [B, T, V]
#pragma unroll
  for (int r = 0; r < 8; ++r) {
    int row = m0 + mo + r;
    int t = row >> 5, b = row & 31;
    obase[r] = b * (TSZ * VSZ) + t * VSZ;
  }

  float m8[8];
#pragma unroll
  for (int r = 0; r < 8; ++r) m8[r] = -3.0e38f;

  for (int nt = wave; nt < VSZ / 16; nt += 8) {
    v8f acc = {};
#pragma unroll
    for (int k = 0; k < 8; ++k) {
      v16h Bt = load_b_tile(Wfc + nt * 16 * HSZ + k * 32, HSZ);
      acc = wmma_f16(A[k], Bt, acc);
    }
    int col = nt * 16 + n;
    float bias = bfc[col];
#pragma unroll
    for (int r = 0; r < 8; ++r) {
      float v = acc[r] + bias;
      out[obase[r] + col] = v;
      m8[r] = fmaxf(m8[r], v);
    }
  }
  // reduce max across the 16 lanes of each half (halves hold disjoint rows)
#pragma unroll
  for (int off = 8; off >= 1; off >>= 1)
#pragma unroll
    for (int r = 0; r < 8; ++r)
      m8[r] = fmaxf(m8[r], __shfl_xor(m8[r], off, 16));
  if (n == 0)
#pragma unroll
    for (int r = 0; r < 8; ++r) red[wave][mo + r] = m8[r];
  __syncthreads();
  if (threadIdx.x < 16) {
    float g = -3.0e38f;
    for (int w = 0; w < 8; ++w) g = fmaxf(g, red[w][threadIdx.x]);
    rowstat[threadIdx.x] = g;
  }
  __syncthreads();

  float rm[8], s8[8];
#pragma unroll
  for (int r = 0; r < 8; ++r) { rm[r] = rowstat[mo + r]; s8[r] = 0.0f; }

  for (int nt = wave; nt < VSZ / 16; nt += 8) {
    int col = nt * 16 + n;
#pragma unroll
    for (int r = 0; r < 8; ++r)
      s8[r] += __expf(out[obase[r] + col] - rm[r]);
  }
#pragma unroll
  for (int off = 8; off >= 1; off >>= 1)
#pragma unroll
    for (int r = 0; r < 8; ++r)
      s8[r] += __shfl_xor(s8[r], off, 16);
  if (n == 0)
#pragma unroll
    for (int r = 0; r < 8; ++r) red[wave][mo + r] = s8[r];
  __syncthreads();
  if (threadIdx.x < 16) {
    float s = 0.0f;
    for (int w = 0; w < 8; ++w) s += red[w][threadIdx.x];
    lse[m0 + threadIdx.x] = rowstat[threadIdx.x] + __logf(s);
  }
}

// ---- out -= lse[row], vectorized float4 (V % 4 == 0 so (b,t) uniform/vec) --
__global__ void k_sub(float* __restrict__ out, const float* __restrict__ lse) {
  int i4 = blockIdx.x * blockDim.x + threadIdx.x;
  int stride = gridDim.x * blockDim.x;
  const int n4 = OUTV / 4;
  for (; i4 < n4; i4 += stride) {
    int i = i4 * 4;
    int b = i / (TSZ * VSZ);
    int t = (i % (TSZ * VSZ)) / VSZ;
    float l = lse[t * BSZ + b];
    float4 v = ((float4*)out)[i4];
    v.x -= l; v.y -= l; v.z -= l; v.w -= l;
    ((float4*)out)[i4] = v;
  }
}

extern "C" void kernel_launch(void* const* d_in, const int* in_sizes, int n_in,
                              void* d_out, int out_size, void* d_ws, size_t ws_size,
                              hipStream_t stream) {
  const int*   target = (const int*)d_in[0];
  const float* ehid   = (const float*)d_in[1];   // [1,B,H]
  const float* emb    = (const float*)d_in[2];
  const float* Wih    = (const float*)d_in[3];
  const float* Whh    = (const float*)d_in[4];
  const float* bih    = (const float*)d_in[5];
  const float* bhh    = (const float*)d_in[6];
  const float* Wfc    = (const float*)d_in[7];
  const float* bfc    = (const float*)d_in[8];
  float* out = (float*)d_out;

  char* ws = (char*)d_ws;
  size_t off = 0;
  auto take = [&](size_t bytes) { void* p = ws + off; off += (bytes + 255) & ~(size_t)255; return p; };

  _Float16* Xf16  = (_Float16*)take((size_t)TB * ESZ * 2);
  _Float16* Wih16 = (_Float16*)take((size_t)G3H * ESZ * 2);
  _Float16* Whh16 = (_Float16*)take((size_t)G3H * HSZ * 2);
  _Float16* Wfc16 = (_Float16*)take((size_t)VSZ * HSZ * 2);
  float*    GX    = (float*)   take((size_t)TB * G3H * 4);
  _Float16* Hall  = (_Float16*)take((size_t)TB * HSZ * 2);
  float*    lse   = (float*)   take((size_t)TB * 4);

  k_cvt_f32_f16<<<256, 256, 0, stream>>>(Wih, Wih16, G3H * ESZ);
  k_cvt_f32_f16<<<256, 256, 0, stream>>>(Whh, Whh16, G3H * HSZ);
  k_cvt_f32_f16<<<4096, 256, 0, stream>>>(Wfc, Wfc16, VSZ * HSZ);
  k_embed<<<TB, ESZ, 0, stream>>>(target, emb, Xf16);
  k_gx<<<TB / 16, 256, 0, stream>>>(Xf16, Wih16, bih, GX);
  k_gru<<<1, 256, 0, stream>>>(ehid, Whh16, bhh, GX, Hall, out + OUTV);
  k_proj<<<TB / 16, 256, 0, stream>>>(Hall, Wfc16, bfc, out, lse);
  k_sub<<<2048, 256, 0, stream>>>(out, lse);
}
